// WindowAttentionMemory_45389214384571
// MI455X (gfx1250) — matrix operand order repaired and checked by reading, not compiled, not run on hardware
//
#include <hip/hip_runtime.h>

typedef _Float16 half8   __attribute__((ext_vector_type(8)));
typedef _Float16 half16  __attribute__((ext_vector_type(16)));
typedef _Float16 half4v  __attribute__((ext_vector_type(4)));
typedef float    float8  __attribute__((ext_vector_type(8)));

#define HID   1024
#define SEQ   2048
#define HEADS 16
#define HDIM  64
#define WIN   256

// ---------------- WMMA helpers (CDNA5 16x16x32 f16, f32 accum) ----------------

// A-matrix 16x32 per-lane layout: lanes 0-15 row M=lane&15 hold K = k0+{0..7} and k0+{16..23};
// lanes 16-31 hold K = k0+{8..15} and k0+{24..31}.
__device__ __forceinline__ half16 ld_a_frag(const _Float16* base, int ldk, int row, int k0, int lane) {
  const int h = (lane >> 4) & 1;
  const _Float16* p = base + (size_t)row * ldk + k0 + 8 * h;
  half8 lo = *(const half8*)p;
  half8 hi = *(const half8*)(p + 16);
  return __builtin_shufflevector(lo, hi, 0,1,2,3,4,5,6,7,8,9,10,11,12,13,14,15);
}

// B-matrix 32x16 per-lane layout: lane holds column N=lane&15, K = k0 + 8*(lane>>4) + {0..7} and k0 + 16 + 8*(lane>>4) + {0..7}.
// "row" is the row index of the k-contiguous storage matrix (W[n,:], K[j,:], vT[d,:]).
__device__ __forceinline__ half16 ld_b_frag(const _Float16* base, int ldk, int row, int k0, int lane) {
  const int h = (lane >> 4) & 1;
  const _Float16* p = base + (size_t)row * ldk + k0 + 8 * h;
  half8 lo = *(const half8*)p;
  half8 hi = *(const half8*)(p + 16);
  return __builtin_shufflevector(lo, hi, 0,1,2,3,4,5,6,7,8,9,10,11,12,13,14,15);
}

__device__ __forceinline__ float8 wmma16(half16 a, half16 b, float8 c) {
  return __builtin_amdgcn_wmma_f32_16x16x32_f16(false, a, false, b, (short)0, c, false, false);
}

// ---------------- LayerNorm -> f16 ----------------
__global__ __launch_bounds__(256) void ln_f16_kernel(const float* __restrict__ x,
                                                     const float* __restrict__ gamma,
                                                     const float* __restrict__ beta,
                                                     _Float16* __restrict__ x16) {
  __shared__ float red[256];
  const int row = blockIdx.x;
  const int tid = threadIdx.x;
  const float* xr = x + (size_t)row * HID;
  float4 v = ((const float4*)xr)[tid];
  float vv[4] = {v.x, v.y, v.z, v.w};

  float s = vv[0] + vv[1] + vv[2] + vv[3];
  red[tid] = s; __syncthreads();
  for (int o = 128; o > 0; o >>= 1) { if (tid < o) red[tid] += red[tid + o]; __syncthreads(); }
  const float mu = red[0] * (1.0f / HID);
  __syncthreads();

  float sq = 0.f;
  #pragma unroll
  for (int k = 0; k < 4; k++) { float d = vv[k] - mu; sq += d * d; }
  red[tid] = sq; __syncthreads();
  for (int o = 128; o > 0; o >>= 1) { if (tid < o) red[tid] += red[tid + o]; __syncthreads(); }
  const float inv = rsqrtf(red[0] * (1.0f / HID) + 1e-12f);

  const int c = tid * 4;
  half4v out;
  #pragma unroll
  for (int k = 0; k < 4; k++) {
    float y = (vv[k] - mu) * inv * gamma[c + k] + beta[c + k];
    out[k] = (_Float16)y;
  }
  ((half4v*)(x16 + (size_t)row * HID))[tid] = out;
}

// ---------------- f32 -> f16 weight cast ----------------
__global__ __launch_bounds__(256) void cast_f16_kernel(const float* __restrict__ src,
                                                       _Float16* __restrict__ dst, int n4) {
  int i = blockIdx.x * 256 + threadIdx.x;
  if (i < n4) {
    float4 f = ((const float4*)src)[i];
    half4v h = {(_Float16)f.x, (_Float16)f.y, (_Float16)f.z, (_Float16)f.w};
    ((half4v*)dst)[i] = h;
  }
}

// ---------------- GEMM: Y = X @ W^T + b, templated epilogue ----------------
// EPI 0: store f16 to [b, head, s, d]       (Q, K)
// EPI 1: store f16 transposed [b, head, d, s] (V)
// EPI 2: store f32  acc + bias + residual   (final out)
template <int EPI>
__global__ __launch_bounds__(256) void gemm_wmma_kernel(const _Float16* __restrict__ A,
                                                        const _Float16* __restrict__ W,
                                                        const float* __restrict__ bias,
                                                        const float* __restrict__ residual,
                                                        _Float16* __restrict__ out16,
                                                        float* __restrict__ out32) {
  const int lane = threadIdx.x & 31;
  const int w    = threadIdx.x >> 5;
  const int wrow = w & 3;
  const int wcol = w >> 2;
  const int M0 = blockIdx.y * 128 + wrow * 32;
  const int N0 = blockIdx.x * 64  + wcol * 32;
  const int ln = lane & 15;

  float8 acc[2][2] = {};
  for (int k0 = 0; k0 < HID; k0 += 32) {
    half16 a0 = ld_a_frag(A, HID, M0 + ln,      k0, lane);
    half16 a1 = ld_a_frag(A, HID, M0 + 16 + ln, k0, lane);
    half16 b0 = ld_b_frag(W, HID, N0 + ln,      k0, lane);
    half16 b1 = ld_b_frag(W, HID, N0 + 16 + ln, k0, lane);
    acc[0][0] = wmma16(a0, b0, acc[0][0]);
    acc[0][1] = wmma16(a0, b1, acc[0][1]);
    acc[1][0] = wmma16(a1, b0, acc[1][0]);
    acc[1][1] = wmma16(a1, b1, acc[1][1]);
  }

  const int h = (lane >> 4) & 1;
  #pragma unroll
  for (int mi = 0; mi < 2; mi++) {
    #pragma unroll
    for (int ni = 0; ni < 2; ni++) {
      #pragma unroll
      for (int r = 0; r < 8; r++) {
        const int m = M0 + 16 * mi + r + 8 * h;   // token row (0..4095)
        const int n = N0 + 16 * ni + ln;          // output feature (0..1023)
        float y = acc[mi][ni][r] + bias[n];
        if (EPI == 0) {
          const int b = m >> 11, s = m & (SEQ - 1), head = n >> 6, d = n & 63;
          out16[(((size_t)(b * HEADS + head) * SEQ) + s) * HDIM + d] = (_Float16)y;
        } else if (EPI == 1) {
          const int b = m >> 11, s = m & (SEQ - 1), head = n >> 6, d = n & 63;
          out16[((size_t)(b * HEADS + head) * HDIM + d) * SEQ + s] = (_Float16)y;
        } else {
          out32[(size_t)m * HID + n] = y + residual[(size_t)m * HID + n];
        }
      }
    }
  }
}

// ---------------- Flash-style sliding-window attention ----------------
// grid: (SEQ/64, B*HEADS), block: 128 (4 waves, 16 queries each)
__global__ __launch_bounds__(128) void attn_wmma_kernel(const _Float16* __restrict__ q16,
                                                        const _Float16* __restrict__ k16,
                                                        const _Float16* __restrict__ vt16,
                                                        _Float16* __restrict__ attn16) {
  __shared__ _Float16 plds[4][16 * 32];
  const int lane = threadIdx.x & 31;
  const int w    = threadIdx.x >> 5;
  const int ln   = lane & 15;
  const int h    = (lane >> 4) & 1;
  const int bh   = blockIdx.y;                 // b*HEADS + head
  const int i0   = blockIdx.x * 64;
  const int qbase = i0 + 16 * w;

  const _Float16* qb = q16  + (size_t)bh * SEQ * HDIM;
  const _Float16* kb = k16  + (size_t)bh * SEQ * HDIM;
  const _Float16* vb = vt16 + (size_t)bh * HDIM * SEQ;

  half16 aq0 = ld_a_frag(qb, HDIM, qbase + ln, 0,  lane);
  half16 aq1 = ld_a_frag(qb, HDIM, qbase + ln, 32, lane);

  float8 O[4] = {};
  float rmax[8], rsum[8];
  #pragma unroll
  for (int r = 0; r < 8; r++) { rmax[r] = -1e30f; rsum[r] = 0.f; }

  int jstart = i0 - WIN;           // i0 multiple of 64 -> multiple of 32
  if (jstart < 0) jstart = 0;

  for (int jt = jstart; jt <= i0 + 63; jt += 32) {
    // ---- scores: two 16-key subtiles, K-reduce over d=64 in two chunks ----
    float8 s0 = {}, s1 = {};
    {
      half16 bk00 = ld_b_frag(kb, HDIM, jt + ln,      0,  lane);
      half16 bk01 = ld_b_frag(kb, HDIM, jt + ln,      32, lane);
      half16 bk10 = ld_b_frag(kb, HDIM, jt + 16 + ln, 0,  lane);
      half16 bk11 = ld_b_frag(kb, HDIM, jt + 16 + ln, 32, lane);
      s0 = wmma16(aq0, bk00, s0);
      s0 = wmma16(aq1, bk01, s0);
      s1 = wmma16(aq0, bk10, s1);
      s1 = wmma16(aq1, bk11, s1);
    }

    // ---- mask + online softmax (per C-frag row r; 16-lane half reductions) ----
    float p0[8], p1[8];
    #pragma unroll
    for (int r = 0; r < 8; r++) {
      const int i  = qbase + r + 8 * h;
      const int j0 = jt + ln;
      const int j1 = jt + 16 + ln;
      const bool a0 = (j0 <= i) && (j0 >= i - (WIN - 1));
      const bool a1 = (j1 <= i) && (j1 >= i - (WIN - 1));
      float v0 = a0 ? s0[r] * 0.125f : -1e30f;
      float v1 = a1 ? s1[r] * 0.125f : -1e30f;

      float t = fmaxf(v0, v1);
      t = fmaxf(t, __shfl_xor(t, 1, 32));
      t = fmaxf(t, __shfl_xor(t, 2, 32));
      t = fmaxf(t, __shfl_xor(t, 4, 32));
      t = fmaxf(t, __shfl_xor(t, 8, 32));

      const float mnew = fmaxf(rmax[r], t);
      const float f    = __expf(rmax[r] - mnew);
      const float e0   = a0 ? __expf(v0 - mnew) : 0.f;
      const float e1   = a1 ? __expf(v1 - mnew) : 0.f;

      float ts = e0 + e1;
      ts += __shfl_xor(ts, 1, 32);
      ts += __shfl_xor(ts, 2, 32);
      ts += __shfl_xor(ts, 4, 32);
      ts += __shfl_xor(ts, 8, 32);

      rsum[r] = rsum[r] * f + ts;
      rmax[r] = mnew;
      p0[r] = e0; p1[r] = e1;
      O[0][r] *= f; O[1][r] *= f; O[2][r] *= f; O[3][r] *= f;
    }

    // ---- P (C-layout) -> LDS row-major -> A-layout fragment ----
    _Float16* pl = &plds[w][0];
    #pragma unroll
    for (int r = 0; r < 8; r++) {
      const int m = r + 8 * h;
      pl[m * 32 + ln]      = (_Float16)p0[r];
      pl[m * 32 + 16 + ln] = (_Float16)p1[r];
    }
    __syncthreads();
    half16 ap = ld_a_frag(pl, 32, ln, 0, lane);
    __syncthreads();

    // ---- O += P @ V  (B from transposed V: rows = d, contiguous along keys) ----
    #pragma unroll
    for (int nt = 0; nt < 4; nt++) {
      half16 bv = ld_b_frag(vb, SEQ, 16 * nt + ln, jt, lane);
      O[nt] = wmma16(ap, bv, O[nt]);
    }
  }

  // ---- finalize: O / rowsum -> attn16 [b, s, head*64 + d] (f16) ----
  const int b = bh >> 4, head = bh & 15;
  #pragma unroll
  for (int nt = 0; nt < 4; nt++) {
    #pragma unroll
    for (int r = 0; r < 8; r++) {
      const int m   = qbase + r + 8 * h;
      const int col = head * HDIM + 16 * nt + ln;
      const float y = O[nt][r] / rsum[r];
      attn16[(size_t)(b * SEQ + m) * HID + col] = (_Float16)y;
    }
  }
}

// ---------------- host launcher ----------------
extern "C" void kernel_launch(void* const* d_in, const int* in_sizes, int n_in,
                              void* d_out, int out_size, void* d_ws, size_t ws_size,
                              hipStream_t stream) {
  const float* hs    = (const float*)d_in[0];
  const float* gamma = (const float*)d_in[1];
  const float* beta  = (const float*)d_in[2];
  const float* Wq    = (const float*)d_in[3];
  const float* bq    = (const float*)d_in[4];
  const float* Wk    = (const float*)d_in[5];
  const float* bk    = (const float*)d_in[6];
  const float* Wv    = (const float*)d_in[7];
  const float* bv    = (const float*)d_in[8];
  const float* Wo    = (const float*)d_in[9];
  const float* bo    = (const float*)d_in[10];
  float* out         = (float*)d_out;

  const size_t MB = 1024 * 1024;
  char* ws = (char*)d_ws;
  _Float16* X16  = (_Float16*)(ws + 0 * MB);   // 4096x1024 f16 = 8 MB
  _Float16* Wq16 = (_Float16*)(ws + 8 * MB);   // 2 MB each
  _Float16* Wk16 = (_Float16*)(ws + 10 * MB);
  _Float16* Wv16 = (_Float16*)(ws + 12 * MB);
  _Float16* Wo16 = (_Float16*)(ws + 14 * MB);
  _Float16* Q16  = (_Float16*)(ws + 16 * MB);  // [b,h,S,64] 8 MB
  _Float16* K16  = (_Float16*)(ws + 24 * MB);  // [b,h,S,64] 8 MB
  _Float16* VT16 = (_Float16*)(ws + 32 * MB);  // [b,h,64,S] 8 MB
  _Float16* A16  = (_Float16*)(ws + 40 * MB);  // attn ctx [4096,1024] 8 MB

  const int M = 2 * SEQ;  // 4096 token rows

  // 1) LayerNorm + cast to f16
  ln_f16_kernel<<<M, 256, 0, stream>>>(hs, gamma, beta, X16);

  // 2) cast weights to f16 (1M elements each -> 256k float4 threads)
  cast_f16_kernel<<<1024, 256, 0, stream>>>(Wq, Wq16, (HID * HID) / 4);
  cast_f16_kernel<<<1024, 256, 0, stream>>>(Wk, Wk16, (HID * HID) / 4);
  cast_f16_kernel<<<1024, 256, 0, stream>>>(Wv, Wv16, (HID * HID) / 4);
  cast_f16_kernel<<<1024, 256, 0, stream>>>(Wo, Wo16, (HID * HID) / 4);

  // 3) projections
  dim3 ggrid(HID / 64, M / 128);
  gemm_wmma_kernel<0><<<ggrid, 256, 0, stream>>>(X16, Wq16, bq, nullptr, Q16, nullptr);
  gemm_wmma_kernel<0><<<ggrid, 256, 0, stream>>>(X16, Wk16, bk, nullptr, K16, nullptr);
  gemm_wmma_kernel<1><<<ggrid, 256, 0, stream>>>(X16, Wv16, bv, nullptr, VT16, nullptr);

  // 4) sliding-window attention
  attn_wmma_kernel<<<dim3(SEQ / 64, 2 * HEADS), 128, 0, stream>>>(Q16, K16, VT16, A16);

  // 5) output projection + residual (f32 out)
  gemm_wmma_kernel<2><<<ggrid, 256, 0, stream>>>(A16, Wo16, bo, hs, nullptr, out);
}
